// Net_72782515798163
// MI455X (gfx1250) — matrix operand order repaired and checked by reading, not compiled, not run on hardware
//
#include <hip/hip_runtime.h>
#include <hip/hip_bf16.h>
#include <math.h>

// ---------------------------------------------------------------------------
// Graph U-Net (GAT + TopKPool + 2-hop augment + SAGE decoder) for gfx1250.
// Adjacency kept as bitmasks (L2-resident); all GEMMs via v_wmma_f32_16x16x32_bf16.
// Packed bit->bf16 expansion (b128 stores), BK=64 stages, ds_load_b128 fragments.
// ---------------------------------------------------------------------------

#define NN      8192
#define DD      128
#define EE      262144
#define KK1     4096
#define KK2     2048
#define W0W     (NN / 32)    // 256 words per A0 row
#define W1W     (KK1 / 32)   // 128 words per A1 row
#define ATT_SLOPE 0.2f
#define ACT_SLOPE 0.01f
#define LP      72           // padded LDS stride (elements): 144B, 16B-aligned

typedef __attribute__((ext_vector_type(16))) __bf16 v16bf;
typedef __attribute__((ext_vector_type(8)))  float  v8f;
typedef __attribute__((ext_vector_type(4)))  float  v4f;

__device__ __forceinline__ float lrelu(float x, float s) { return x > 0.f ? x : s * x; }

// two contiguous 16-byte LDS runs -> one v16bf fragment (2x ds_load_b128)
__device__ __forceinline__ v16bf frag2(const __bf16* p0, const __bf16* p1)
{
    union { v16bf v; v4f q[2]; } u;
    u.q[0] = *(const v4f*)p0;
    u.q[1] = *(const v4f*)p1;
    return u.v;
}

// expand 8 adjacency bits -> 8 bf16 {0,1} packed in a uint4 (one b128 store)
__device__ __forceinline__ uint4 expand8(unsigned b)
{
    uint4 o;
    o.x = (b & 1u   ? 0x3F80u : 0u) | (b & 2u   ? 0x3F800000u : 0u);
    o.y = (b & 4u   ? 0x3F80u : 0u) | (b & 8u   ? 0x3F800000u : 0u);
    o.z = (b & 16u  ? 0x3F80u : 0u) | (b & 32u  ? 0x3F800000u : 0u);
    o.w = (b & 64u  ? 0x3F80u : 0u) | (b & 128u ? 0x3F800000u : 0u);
    return o;
}

// ---------------------------------------------------------------------------
// Generic tiled GEMM: C[M,N] = A[M,K] @ B[K,N] (+ beta*C), bf16 WMMA, f32 accum.
// A_BITS: A is a row-major bitmask [m][k], lda = words/row.
// B_BITS: B is given TRANSPOSED as a bitmask [n][k], ldb = words/row.
// 128 threads = 4 waves, 64x64 C tile, BK=64 (2 WMMA k-steps per barrier).
// A staged [row][k] stride LP, B staged [col][k] stride LP.
// ---------------------------------------------------------------------------
template <int A_BITS, int B_BITS>
__global__ __launch_bounds__(128) void gemm64x64_bf16_wmma(
    const void* __restrict__ Ap, const void* __restrict__ Bp,
    float* __restrict__ C, int M, int N, int K,
    int lda, int ldb, int ldc, float beta)
{
    __shared__ __bf16 As[64 * LP];   // [row][k 0..63]
    __shared__ __bf16 Bs[64 * LP];   // [col][k 0..63]

    const int tid = threadIdx.x;
    const int m0 = blockIdx.x * 64, n0 = blockIdx.y * 64;

    v8f acc[4];
    for (int q = 0; q < 4; ++q)
        for (int e = 0; e < 8; ++e) acc[q][e] = 0.f;

    for (int k0 = 0; k0 < K; k0 += 64) {
        // ---- stage A tile (64 rows x 64 K) ----
        if (A_BITS) {
            const unsigned* Ab = (const unsigned*)Ap;
            for (int q = tid; q < 512; q += 128) {          // 64 rows x 8 byte-groups
                int r = q >> 3, g = q & 7;
                unsigned w = Ab[(size_t)(m0 + r) * lda + (k0 >> 5) + (g >> 2)];
                *(uint4*)&As[r * LP + g * 8] = expand8((w >> ((g & 3) * 8)) & 0xFFu);
            }
        } else {
            const float* Af = (const float*)Ap;
            for (int q = tid; q < 1024; q += 128) {          // 64 rows x 16 float4
                int r = q >> 4, c4 = (q & 15) * 4;
                v4f v = *(const v4f*)&Af[(size_t)(m0 + r) * lda + k0 + c4];
                union { __bf16 h[4]; uint2 u; } wv;
#pragma unroll
                for (int i = 0; i < 4; ++i) wv.h[i] = (__bf16)v[i];
                *(uint2*)&As[r * LP + c4] = wv.u;
            }
            if (k0 + 64 < K)
                __builtin_prefetch(&Af[(size_t)(m0 + (tid >> 1)) * lda + k0 + 64 + (tid & 1) * 32], 0, 0);
        }
        // ---- stage B tile (64 K x 64 cols) into [col][k] ----
        if (B_BITS) {
            const unsigned* Bt = (const unsigned*)Bp;        // transposed bits [n][k]
            for (int q = tid; q < 512; q += 128) {           // 64 cols x 8 byte-groups
                int c = q >> 3, g = q & 7;
                unsigned w = Bt[(size_t)(n0 + c) * ldb + (k0 >> 5) + (g >> 2)];
                *(uint4*)&Bs[c * LP + g * 8] = expand8((w >> ((g & 3) * 8)) & 0xFFu);
            }
        } else {
            const float* Bf = (const float*)Bp;
            for (int q = tid; q < 1024; q += 128) {          // 64 K x 16 float4
                int r = q >> 4, c4 = (q & 15) * 4;
                v4f v = *(const v4f*)&Bf[(size_t)(k0 + r) * ldb + n0 + c4];
#pragma unroll
                for (int i = 0; i < 4; ++i) Bs[(c4 + i) * LP + r] = (__bf16)v[i];
            }
            if (k0 + 64 < K)
                __builtin_prefetch(&Bf[(size_t)(k0 + 64 + (tid >> 1)) * ldb + n0 + (tid & 1) * 32], 0, 0);
        }
        __syncthreads();

        const int wave = tid >> 5, lane = tid & 31;
        const int half = lane >> 4, lr = lane & 15;

#pragma unroll
        for (int s = 0; s < 2; ++s) {
            const __bf16* ap = &As[(wave * 16 + lr) * LP + s * 32 + half * 8];
            v16bf a = frag2(ap, ap + 16);
#pragma unroll
            for (int nt = 0; nt < 4; ++nt) {
                const __bf16* bp = &Bs[(nt * 16 + lr) * LP + s * 32 + half * 8];
                v16bf b = frag2(bp, bp + 16);
                acc[nt] = __builtin_amdgcn_wmma_f32_16x16x32_bf16(
                    false, a, false, b, (short)0, acc[nt], false, false);
            }
        }
        __syncthreads();
    }

    const int wave = tid >> 5, lane = tid & 31;
    const int half = lane >> 4, lr = lane & 15;
    for (int nt = 0; nt < 4; ++nt)
        for (int r = 0; r < 8; ++r) {
            size_t idx = (size_t)(m0 + wave * 16 + half * 8 + r) * ldc + (n0 + nt * 16 + lr);
            float o = acc[nt][r];
            if (beta != 0.f) o += beta * C[idx];
            C[idx] = o;
        }
}

// ---------------------------------------------------------------------------
// 32x32 bit-tile transpose: dst bit(j,i) = src bit(i,j). Square n x n bits.
// 256 threads handle 8 tiles; grid must cover (n/32)^2 tiles exactly.
// ---------------------------------------------------------------------------
__global__ __launch_bounds__(256) void transpose_bits(
    const unsigned* __restrict__ src, unsigned* __restrict__ dst, int n, int words)
{
    __shared__ unsigned tile[8][32];
    const int grp = threadIdx.x >> 5, t = threadIdx.x & 31;
    const int ntile = n >> 5;
    const int tau = blockIdx.x * 8 + grp;
    const int tr = tau / ntile, tc = tau % ntile;
    tile[grp][t] = src[(size_t)(tr * 32 + t) * words + tc];
    __syncthreads();
    unsigned o = 0;
#pragma unroll
    for (int b = 0; b < 32; ++b)
        o |= ((tile[grp][b] >> t) & 1u) << (unsigned)b;
    dst[(size_t)(tc * 32 + t) * words + tr] = o;
}

// ---------------------------------------------------------------------------
// GAT stats: per-row online (max, sum-exp) over bitmask-selected entries.
// ---------------------------------------------------------------------------
__global__ __launch_bounds__(256) void gat_stats(
    const unsigned* __restrict__ mask, int words,
    const float* __restrict__ ssrc, const float* __restrict__ sdst,
    float* __restrict__ mrow, float* __restrict__ zrow, float slope)
{
    const int row = blockIdx.x, tid = threadIdx.x;
    const float si = ssrc[row];
    float m = -3.0e38f, s = 0.f;
    for (int w = tid; w < words; w += 256) {
        unsigned bits = mask[(size_t)row * words + w];
        int base = w * 32;
        while (bits) {
            int b = __ffs(bits) - 1;
            bits &= bits - 1;
            float ev = lrelu(si + sdst[base + b], slope);
            if (ev > m) { s *= __expf(m - ev); m = ev; }
            s += __expf(ev - m);
        }
    }
    __shared__ float sm[256], ss[256];
    sm[tid] = m; ss[tid] = s;
    __syncthreads();
    for (int off = 128; off > 0; off >>= 1) {
        if (tid < off) {
            float m2 = sm[tid + off], s2 = ss[tid + off];
            float mo = sm[tid], so = ss[tid];
            float mm = fmaxf(mo, m2);
            sm[tid] = mm;
            ss[tid] = so * __expf(mo - mm) + s2 * __expf(m2 - mm);
        }
        __syncthreads();
    }
    if (tid == 0) { mrow[row] = sm[0]; zrow[row] = ss[0]; }
}

// ---------------------------------------------------------------------------
// Fused attention A*V: out[i,:] = sum_j alpha_ij * h[j,:], flash-style.
// 512 threads = 16 waves; 64 rows x 128 cols; j in chunks of 64 (2 k-steps).
// Ps: [row][j] stride LP; Hs: [col][j] stride LP.
// ---------------------------------------------------------------------------
__global__ __launch_bounds__(512) void gat_av_wmma(
    const unsigned* __restrict__ mask, int words,
    const float* __restrict__ ssrc, const float* __restrict__ sdst,
    const float* __restrict__ mrow, const float* __restrict__ zrow,
    const float* __restrict__ h, float* __restrict__ out,
    int ncols, float slope)
{
    __shared__ __bf16 Ps[64 * LP];
    __shared__ __bf16 Hs[128 * LP];
    const int tid = threadIdx.x;
    const int i0 = blockIdx.x * 64;

    v8f acc[2];
    for (int q = 0; q < 2; ++q)
        for (int e = 0; e < 8; ++e) acc[q][e] = 0.f;

    for (int j0 = 0; j0 < ncols; j0 += 64) {
        // alpha tile 64x64
        for (int e = tid; e < 64 * 64; e += 512) {
            int r = e >> 6, c = e & 63;
            int i = i0 + r, j = j0 + c;
            unsigned bit = (mask[(size_t)i * words + (j >> 5)] >> (j & 31)) & 1u;
            float p = 0.f;
            if (bit) {
                float ev = lrelu(ssrc[i] + sdst[j], slope);
                p = __expf(ev - mrow[i]) / zrow[i];
            }
            Ps[r * LP + c] = (__bf16)p;
        }
        // h tile 64x128 -> [col][j]
        for (int q = tid; q < 2048; q += 512) {              // 64 rows x 32 float4
            int r = q >> 5, c4 = (q & 31) * 4;
            v4f v = *(const v4f*)&h[(size_t)(j0 + r) * 128 + c4];
#pragma unroll
            for (int i = 0; i < 4; ++i) Hs[(c4 + i) * LP + r] = (__bf16)v[i];
        }
        if (j0 + 64 < ncols)
            __builtin_prefetch(&h[(size_t)(j0 + 64 + (tid >> 3)) * 128 + (tid & 7) * 16], 0, 0);
        __syncthreads();

        const int wave = tid >> 5, lane = tid & 31;
        const int half = lane >> 4, lr = lane & 15;
        const int wr = wave >> 2, wc = wave & 3;

#pragma unroll
        for (int s = 0; s < 2; ++s) {
            const __bf16* ap = &Ps[(wr * 16 + lr) * LP + s * 32 + half * 8];
            v16bf a = frag2(ap, ap + 16);
#pragma unroll
            for (int nt = 0; nt < 2; ++nt) {
                const __bf16* bp = &Hs[(wc * 32 + nt * 16 + lr) * LP + s * 32 + half * 8];
                v16bf b = frag2(bp, bp + 16);
                acc[nt] = __builtin_amdgcn_wmma_f32_16x16x32_bf16(
                    false, a, false, b, (short)0, acc[nt], false, false);
            }
        }
        __syncthreads();
    }

    const int wave = tid >> 5, lane = tid & 31;
    const int half = lane >> 4, lr = lane & 15;
    const int wr = wave >> 2, wc = wave & 3;
    for (int nt = 0; nt < 2; ++nt)
        for (int r = 0; r < 8; ++r)
            out[(size_t)(i0 + wr * 16 + half * 8 + r) * 128 + wc * 32 + nt * 16 + lr] = acc[nt][r];
}

// ---------------------------------------------------------------------------
// Small utility kernels
// ---------------------------------------------------------------------------
__global__ void build_adj_bits(const int* __restrict__ ei, unsigned* __restrict__ bits,
                               int words, int nE)
{
    int e = blockIdx.x * blockDim.x + threadIdx.x;
    if (e < nE) {
        int r = ei[e], c = ei[nE + e];
        atomicOr(&bits[(size_t)r * words + (c >> 5)], 1u << (c & 31));
    }
}

__global__ void add_self_loops(unsigned* __restrict__ bits, int words, int n)
{
    int i = blockIdx.x * blockDim.x + threadIdx.x;
    if (i < n) atomicOr(&bits[(size_t)i * words + (i >> 5)], 1u << (i & 31));
}

// row dot-product with vector (D = 128); normalize => divide by ||v||
__global__ __launch_bounds__(128) void matvec128(
    const float* __restrict__ X, const float* __restrict__ v,
    float* __restrict__ out, int normalize)
{
    int row = blockIdx.x, tid = threadIdx.x;
    float pv = v[tid];
    __shared__ float sd[128], sp[128];
    sd[tid] = X[(size_t)row * 128 + tid] * pv;
    sp[tid] = pv * pv;
    __syncthreads();
    for (int off = 64; off > 0; off >>= 1) {
        if (tid < off) { sd[tid] += sd[tid + off]; sp[tid] += sp[tid + off]; }
        __syncthreads();
    }
    if (tid == 0) out[row] = normalize ? sd[0] / sqrtf(sp[0]) : sd[0];
}

// single-workgroup bitonic sort, descending by value (ties: ascending index)
template <int M>
__global__ __launch_bounds__(1024) void topk_sort(
    const float* __restrict__ score, int* __restrict__ perm,
    float* __restrict__ vals, int k)
{
    __shared__ float v[M];
    __shared__ int ix[M];
    int tid = threadIdx.x;
    for (int e = tid; e < M; e += 1024) { v[e] = score[e]; ix[e] = e; }
    __syncthreads();
    for (int kk = 2; kk <= M; kk <<= 1)
        for (int j = kk >> 1; j > 0; j >>= 1) {
            for (int t = tid; t < M; t += 1024) {
                int l = t ^ j;
                if (l > t) {
                    bool dir = (t & kk) == 0;
                    float va = v[t], vb = v[l];
                    int ia = ix[t], ib = ix[l];
                    bool tFirst = (va > vb) || (va == vb && ia < ib);
                    if (dir ? !tFirst : tFirst) {
                        v[t] = vb; v[l] = va; ix[t] = ib; ix[l] = ia;
                    }
                }
            }
            __syncthreads();
        }
    for (int e = tid; e < k; e += 1024) { perm[e] = ix[e]; vals[e] = v[e]; }
}

// xp[i,:] = leaky_relu(X[perm[i],:], slope) * tanh(vals[i])
__global__ void pool_gather(const float* __restrict__ X, const int* __restrict__ perm,
                            const float* __restrict__ vals, float* __restrict__ out,
                            int k, float slope)
{
    int t = blockIdx.x * blockDim.x + threadIdx.x;
    if (t < k * 128) {
        int i = t >> 7, d = t & 127;
        out[t] = lrelu(X[(size_t)perm[i] * 128 + d], slope) * tanhf(vals[i]);
    }
}

// dst bitmask = src[perm][:, perm]
__global__ void gather_adj_bits(const unsigned* __restrict__ src, int wsrc,
                                const int* __restrict__ perm,
                                unsigned* __restrict__ dst, int wdst, int k)
{
    int t = blockIdx.x * blockDim.x + threadIdx.x;
    if (t < k * wdst) {
        int i = t / wdst, w = t % wdst;
        int rs = perm[i];
        unsigned acc = 0;
        for (int b = 0; b < 32; ++b) {
            int c = perm[w * 32 + b];
            acc |= ((src[(size_t)rs * wsrc + (c >> 5)] >> (c & 31)) & 1u) << (unsigned)b;
        }
        dst[t] = acc;
    }
}

// bitmask[i][w] from f32 counts: bit = (C[i][j] > 0.5)
__global__ void threshold_bits(const float* __restrict__ C, unsigned* __restrict__ bits,
                               int n, int words)
{
    int t = blockIdx.x * blockDim.x + threadIdx.x;
    if (t < n * words) {
        int i = t / words, w = t % words;
        unsigned acc = 0;
        for (int b = 0; b < 32; ++b)
            acc |= (C[(size_t)i * n + w * 32 + b] > 0.5f ? 1u : 0u) << (unsigned)b;
        bits[t] = acc;
    }
}

__global__ void row_deg(const unsigned* __restrict__ bits, int words,
                        float* __restrict__ deg, int n)
{
    int i = blockIdx.x * blockDim.x + threadIdx.x;
    if (i < n) {
        int c = 0;
        for (int w = 0; w < words; ++w) c += __popc(bits[(size_t)i * words + w]);
        deg[i] = c < 1 ? 1.f : (float)c;
    }
}

__global__ void row_scale(float* __restrict__ X, const float* __restrict__ deg, int n)
{
    int t = blockIdx.x * blockDim.x + threadIdx.x;
    if (t < n * 128) X[t] /= deg[t >> 7];
}

__global__ void bias_act(float* __restrict__ X, const float* __restrict__ b, int n, int relu)
{
    int t = blockIdx.x * blockDim.x + threadIdx.x;
    if (t < n * 128) {
        float v = X[t] + b[t & 127];
        X[t] = relu ? fmaxf(v, 0.f) : v;
    }
}

__global__ void scatter_rows(const float* __restrict__ src, const int* __restrict__ perm,
                             float* __restrict__ dst, int k)
{
    int t = blockIdx.x * blockDim.x + threadIdx.x;
    if (t < k * 128) {
        int i = t >> 7, d = t & 127;
        dst[(size_t)perm[i] * 128 + d] = src[t];
    }
}

__global__ void emit_A2(const unsigned* __restrict__ bits, int words,
                        const int* __restrict__ perm, float* __restrict__ A2, int k)
{
    int t = blockIdx.x * blockDim.x + threadIdx.x;
    if (t < k * k) {
        int i = t / k, j = t % k;
        int r = perm[i], c = perm[j];
        A2[t] = (float)((bits[(size_t)r * words + (c >> 5)] >> (c & 31)) & 1u);
    }
}

__global__ void emit_batch(const int* __restrict__ batch, const int* __restrict__ perm0,
                           const int* __restrict__ perm1, float* __restrict__ out, int k)
{
    int t = blockIdx.x * blockDim.x + threadIdx.x;
    if (t < k) out[t] = (float)batch[perm0[perm1[t]]];
}

// ---------------------------------------------------------------------------
// Host orchestration
// ---------------------------------------------------------------------------
extern "C" void kernel_launch(void* const* d_in, const int* in_sizes, int n_in,
                              void* d_out, int out_size, void* d_ws, size_t ws_size,
                              hipStream_t stream)
{
    (void)in_sizes; (void)n_in; (void)out_size; (void)ws_size;

    const float* x     = (const float*)d_in[0];
    const float* W0    = (const float*)d_in[1];
    const float* a0s   = (const float*)d_in[2];
    const float* a0d   = (const float*)d_in[3];
    const float* p0    = (const float*)d_in[4];
    const float* W1    = (const float*)d_in[5];
    const float* a1s   = (const float*)d_in[6];
    const float* a1d   = (const float*)d_in[7];
    const float* p1    = (const float*)d_in[8];
    const float* Wr0   = (const float*)d_in[9];
    const float* Wn0   = (const float*)d_in[10];
    const float* bu0   = (const float*)d_in[11];
    const float* Wr1   = (const float*)d_in[12];
    const float* Wn1   = (const float*)d_in[13];
    const float* bu1   = (const float*)d_in[14];
    const int*   ei    = (const int*)d_in[15];
    const int*   batch = (const int*)d_in[16];

    float* out = (float*)d_out;
    float* z_out      = out;                                        // N*D
    float* latent_out = out + (size_t)NN * DD;                      // K2*D
    float* A2_out     = latent_out + (size_t)KK2 * DD;              // K2*K2
    float* b_out      = A2_out + (size_t)KK2 * KK2;                 // K2

    // workspace carve-out (256B aligned)
    char* base = (char*)d_ws;
    size_t cur = 0;
    auto alloc = [&](size_t bytes) -> char* {
        char* p = base + cur;
        cur += (bytes + 255) & ~(size_t)255;
        return p;
    };
    unsigned* A0bits  = (unsigned*)alloc((size_t)NN * W0W * 4);
    float*    h0      = (float*)alloc((size_t)NN * DD * 4);
    float*    s0src   = (float*)alloc(NN * 4);
    float*    s0dst   = (float*)alloc(NN * 4);
    float*    m0      = (float*)alloc(NN * 4);
    float*    z0      = (float*)alloc(NN * 4);
    float*    out0    = (float*)alloc((size_t)NN * DD * 4);
    float*    score0  = (float*)alloc(NN * 4);
    int*      perm0   = (int*)alloc(KK1 * 4);
    float*    vals0   = (float*)alloc(KK1 * 4);
    float*    xp0     = (float*)alloc((size_t)KK1 * DD * 4);
    unsigned* A1bits  = (unsigned*)alloc((size_t)KK1 * W1W * 4);
    unsigned* A1bitsT = (unsigned*)alloc((size_t)KK1 * W1W * 4);
    float*    A1cnt   = (float*)alloc((size_t)KK1 * KK1 * 4);
    unsigned* A1bits2 = (unsigned*)alloc((size_t)KK1 * W1W * 4);
    float*    h1      = (float*)alloc((size_t)KK1 * DD * 4);
    float*    s1src   = (float*)alloc(KK1 * 4);
    float*    s1dst   = (float*)alloc(KK1 * 4);
    float*    m1      = (float*)alloc(KK1 * 4);
    float*    z1      = (float*)alloc(KK1 * 4);
    float*    out1    = (float*)alloc((size_t)KK1 * DD * 4);
    float*    score1  = (float*)alloc(KK1 * 4);
    int*      perm1   = (int*)alloc(KK2 * 4);
    float*    vals1   = (float*)alloc(KK2 * 4);
    float*    up1     = (float*)alloc((size_t)KK1 * DD * 4);
    float*    t1      = (float*)alloc((size_t)KK1 * DD * 4);
    float*    z1d     = (float*)alloc((size_t)KK1 * DD * 4);
    float*    deg1    = (float*)alloc(KK1 * 4);
    float*    up0     = (float*)alloc((size_t)NN * DD * 4);
    float*    t0      = (float*)alloc((size_t)NN * DD * 4);
    float*    deg0    = (float*)alloc(NN * 4);

    // ---- adjacency A0 (bitmask, with self loops) ----
    hipMemsetAsync(A0bits, 0, (size_t)NN * W0W * 4, stream);
    build_adj_bits<<<(EE + 255) / 256, 256, 0, stream>>>(ei, A0bits, W0W, EE);
    add_self_loops<<<(NN + 255) / 256, 256, 0, stream>>>(A0bits, W0W, NN);

    // ---- encoder level 0: GAT ----
    gemm64x64_bf16_wmma<0, 0><<<dim3(NN / 64, DD / 64), 128, 0, stream>>>(
        x, W0, h0, NN, DD, DD, DD, DD, DD, 0.f);
    matvec128<<<NN, 128, 0, stream>>>(h0, a0s, s0src, 0);
    matvec128<<<NN, 128, 0, stream>>>(h0, a0d, s0dst, 0);
    gat_stats<<<NN, 256, 0, stream>>>(A0bits, W0W, s0src, s0dst, m0, z0, ATT_SLOPE);
    gat_av_wmma<<<NN / 64, 512, 0, stream>>>(A0bits, W0W, s0src, s0dst, m0, z0, h0,
                                             out0, NN, ATT_SLOPE);
    // ---- pool 0 ----
    matvec128<<<NN, 128, 0, stream>>>(out0, p0, score0, 1);
    topk_sort<NN><<<1, 1024, 0, stream>>>(score0, perm0, vals0, KK1);
    pool_gather<<<(KK1 * DD + 255) / 256, 256, 0, stream>>>(out0, perm0, vals0, xp0,
                                                            KK1, ACT_SLOPE);
    gather_adj_bits<<<(KK1 * W1W + 255) / 256, 256, 0, stream>>>(
        A0bits, W0W, perm0, A1bits, W1W, KK1);
    // augment: A1 = (A1 @ A1) > 0   (binary matmul: exact in bf16 WMMA)
    transpose_bits<<<(KK1 / 32) * (KK1 / 32) / 8, 256, 0, stream>>>(
        A1bits, A1bitsT, KK1, W1W);
    gemm64x64_bf16_wmma<1, 1><<<dim3(KK1 / 64, KK1 / 64), 128, 0, stream>>>(
        A1bits, A1bitsT, A1cnt, KK1, KK1, KK1, W1W, W1W, KK1, 0.f);
    threshold_bits<<<(KK1 * W1W + 255) / 256, 256, 0, stream>>>(A1cnt, A1bits2, KK1, W1W);

    // ---- encoder level 1: GAT ----
    gemm64x64_bf16_wmma<0, 0><<<dim3(KK1 / 64, DD / 64), 128, 0, stream>>>(
        xp0, W1, h1, KK1, DD, DD, DD, DD, DD, 0.f);
    matvec128<<<KK1, 128, 0, stream>>>(h1, a1s, s1src, 0);
    matvec128<<<KK1, 128, 0, stream>>>(h1, a1d, s1dst, 0);
    gat_stats<<<KK1, 256, 0, stream>>>(A1bits2, W1W, s1src, s1dst, m1, z1, ATT_SLOPE);
    gat_av_wmma<<<KK1 / 64, 512, 0, stream>>>(A1bits2, W1W, s1src, s1dst, m1, z1, h1,
                                              out1, KK1, ATT_SLOPE);
    // ---- pool 1 ----
    matvec128<<<KK1, 128, 0, stream>>>(out1, p1, score1, 1);
    topk_sort<KK1><<<1, 1024, 0, stream>>>(score1, perm1, vals1, KK2);
    pool_gather<<<(KK2 * DD + 255) / 256, 256, 0, stream>>>(out1, perm1, vals1,
                                                            latent_out, KK2, ACT_SLOPE);
    emit_A2<<<(KK2 * KK2 + 255) / 256, 256, 0, stream>>>(A1bits2, W1W, perm1, A2_out, KK2);
    emit_batch<<<(KK2 + 255) / 256, 256, 0, stream>>>(batch, perm0, perm1, b_out, KK2);

    // ---- decoder: SAGE on A1 (augmented), then SAGE on A0 ----
    hipMemsetAsync(up1, 0, (size_t)KK1 * DD * 4, stream);
    scatter_rows<<<(KK2 * DD + 255) / 256, 256, 0, stream>>>(latent_out, perm1, up1, KK2);
    gemm64x64_bf16_wmma<1, 0><<<dim3(KK1 / 64, DD / 64), 128, 0, stream>>>(
        A1bits2, up1, t1, KK1, DD, KK1, W1W, DD, DD, 0.f);
    row_deg<<<(KK1 + 255) / 256, 256, 0, stream>>>(A1bits2, W1W, deg1, KK1);
    row_scale<<<(KK1 * DD + 255) / 256, 256, 0, stream>>>(t1, deg1, KK1);
    gemm64x64_bf16_wmma<0, 0><<<dim3(KK1 / 64, DD / 64), 128, 0, stream>>>(
        up1, Wr0, z1d, KK1, DD, DD, DD, DD, DD, 0.f);
    gemm64x64_bf16_wmma<0, 0><<<dim3(KK1 / 64, DD / 64), 128, 0, stream>>>(
        t1, Wn0, z1d, KK1, DD, DD, DD, DD, DD, 1.f);
    bias_act<<<(KK1 * DD + 255) / 256, 256, 0, stream>>>(z1d, bu0, KK1, 1);

    hipMemsetAsync(up0, 0, (size_t)NN * DD * 4, stream);
    scatter_rows<<<(KK1 * DD + 255) / 256, 256, 0, stream>>>(z1d, perm0, up0, KK1);
    gemm64x64_bf16_wmma<1, 0><<<dim3(NN / 64, DD / 64), 128, 0, stream>>>(
        A0bits, up0, t0, NN, DD, NN, W0W, DD, DD, 0.f);
    row_deg<<<(NN + 255) / 256, 256, 0, stream>>>(A0bits, W0W, deg0, NN);
    row_scale<<<(NN * DD + 255) / 256, 256, 0, stream>>>(t0, deg0, NN);
    gemm64x64_bf16_wmma<0, 0><<<dim3(NN / 64, DD / 64), 128, 0, stream>>>(
        up0, Wr1, z_out, NN, DD, DD, DD, DD, DD, 0.f);
    gemm64x64_bf16_wmma<0, 0><<<dim3(NN / 64, DD / 64), 128, 0, stream>>>(
        t0, Wn1, z_out, NN, DD, DD, DD, DD, DD, 1.f);
    bias_act<<<(NN * DD + 255) / 256, 256, 0, stream>>>(z_out, bu1, NN, 0);
}
